// ConformerBlock_50903952392464
// MI455X (gfx1250) — matrix operand order repaired and checked by reading, not compiled, not run on hardware
//
#include <hip/hip_runtime.h>
#include <math.h>

typedef __attribute__((ext_vector_type(16))) _Float16 v16h;
typedef __attribute__((ext_vector_type(8)))  _Float16 v8h;
typedef __attribute__((ext_vector_type(8)))  float    v8f;

#define WMMA_F32_F16(a,b,c) \
  __builtin_amdgcn_wmma_f32_16x16x32_f16(false,(a),false,(b),(short)0,(c),false,false)

static constexpr int CB  = 2;
static constexpr int CD  = 512;
static constexpr int CT  = 2048;
static constexpr int CH  = 8;
static constexpr int CHD = 64;

// ---------------------------------------------------------------------------
// WMMA GEMM:  C[m,n] = res[m,n] + alpha * act( sum_k A[m,k]*B[k,n] + bias[n] )
// A row-major MxK contiguous. B strided (bK,bN); BTRANS=0 -> contiguous in n
// (bN==1), BTRANS=1 -> contiguous in k (bK==1).  Template-specialized so the
// K-loop has zero control flow.  Block: 256 thr = 8 waves -> 128x64 tile;
// B tile staged in LDS as f16 shared by all 8 waves; fragments via ds_load_b128.
// ---------------------------------------------------------------------------
template <int BTRANS>
__global__ void __launch_bounds__(256)
wmma_gemm_kernel(const float* __restrict__ A, const float* __restrict__ Bm,
                 long bK, long bN,
                 const float* __restrict__ bias, const float* __restrict__ res,
                 float* __restrict__ C, int M, int N, int K,
                 float alpha, int act)
{
  __shared__ _Float16 Blds[64][40];   // [n][k], stride 80B (16B aligned, odd-dword banks)
  const int tid  = threadIdx.x;
  const int lane = tid & 31;
  const int wave = tid >> 5;
  const int m0 = (blockIdx.y * 8 + wave) * 16;
  const int n0 = blockIdx.x * 64;
  const int rw  = lane & 15;
  const int grp = lane >> 4;
  const int kbA = grp * 8;       // A-fragment K base (ISA layout)
  const int kbB = grp * 8;       // B-fragment K base (ISA layout)

  const float* Arow = A + (long)(m0 + rw) * K;
  v8f acc[4] = {};

  for (int k0 = 0; k0 < K; k0 += 32) {
    // ---- stage B tile (32 x 64) -> LDS f16, coalesced b128 global loads ----
    if constexpr (BTRANS == 0) {
#pragma unroll
      for (int it = 0; it < 2; ++it) {
        const int idx = tid + it * 256;        // [0,512) float4s
        const int k = idx >> 4;                // 0..31
        const int n = (idx & 15) * 4;          // 0..60
        const float4 v = *(const float4*)(Bm + (long)(k0 + k) * bK + (n0 + n));
        Blds[n + 0][k] = (_Float16)v.x;
        Blds[n + 1][k] = (_Float16)v.y;
        Blds[n + 2][k] = (_Float16)v.z;
        Blds[n + 3][k] = (_Float16)v.w;
      }
    } else {
#pragma unroll
      for (int it = 0; it < 2; ++it) {
        const int idx = tid + it * 256;
        const int n = idx >> 3;                // 0..63
        const int k = (idx & 7) * 4;           // 0..28
        const float4 v = *(const float4*)(Bm + (long)(n0 + n) * bN + (k0 + k));
        Blds[n][k + 0] = (_Float16)v.x;
        Blds[n][k + 1] = (_Float16)v.y;
        Blds[n][k + 2] = (_Float16)v.z;
        Blds[n][k + 3] = (_Float16)v.w;
      }
    }
    __syncthreads();

    // ---- A fragment: 4 x global b128 + packed cvt (+ prefetch next tile) ----
    if (k0 + 32 < K) __builtin_prefetch(Arow + k0 + 32, 0, 3);  // global_prefetch_b8
    v16h af;
    {
      const float4* a0 = (const float4*)(Arow + k0 + kbA);
      const float4* a1 = (const float4*)(Arow + k0 + 16 + kbA);
      const float4 x0 = a0[0], x1 = a0[1], x2 = a1[0], x3 = a1[1];
      af[0]=(_Float16)x0.x; af[1]=(_Float16)x0.y; af[2]=(_Float16)x0.z; af[3]=(_Float16)x0.w;
      af[4]=(_Float16)x1.x; af[5]=(_Float16)x1.y; af[6]=(_Float16)x1.z; af[7]=(_Float16)x1.w;
      af[8]=(_Float16)x2.x; af[9]=(_Float16)x2.y; af[10]=(_Float16)x2.z; af[11]=(_Float16)x2.w;
      af[12]=(_Float16)x3.x; af[13]=(_Float16)x3.y; af[14]=(_Float16)x3.z; af[15]=(_Float16)x3.w;
    }

    // ---- B fragments from LDS (2 x ds_load_b128 each) + 4 WMMAs ----
#pragma unroll
    for (int t = 0; t < 4; ++t) {
      const _Float16* bp = &Blds[t * 16 + rw][kbB];
      const v8h lo = *(const v8h*)bp;
      const v8h hi = *(const v8h*)(bp + 16);
      v16h bf;
#pragma unroll
      for (int j = 0; j < 8; ++j) { bf[j] = lo[j]; bf[j + 8] = hi[j]; }
      acc[t] = WMMA_F32_F16(af, bf, acc[t]);
    }
    __syncthreads();
  }

  // ---- epilogue (C layout: vgpr r -> row m0+r+8*grp, col n0+t*16+rw) ----
#pragma unroll
  for (int r = 0; r < 8; ++r) {
    const long cb = (long)(m0 + r + grp * 8) * N;
#pragma unroll
    for (int t = 0; t < 4; ++t) {
      const int n = n0 + t * 16 + rw;
      float v = acc[t][r];
      if (bias) v += bias[n];
      if (act == 1) v = 0.5f * v * (1.0f + erff(v * 0.70710678118654752f));
      v *= alpha;
      if (res) v += res[cb + n];
      C[cb + n] = v;
    }
  }
}

// ---------------------------------------------------------------------------
// Flash attention with gated relative bias.  qkv packed (B*T,1536).
// Block = 4 waves sharing (b,h); K/V chunks staged in LDS once per 32-key
// step and shared by all 4 waves; per-wave 16 query rows, online softmax.
// ---------------------------------------------------------------------------
__global__ void __launch_bounds__(128)
flash_attn_kernel(const float* __restrict__ qkv, const float* __restrict__ mmult,
                  const float* __restrict__ tbl, float* __restrict__ out)
{
  __shared__ _Float16 Klds[32][72];        // [key][d], 144B stride
  __shared__ _Float16 Vlds[64][40];        // [d][key], 80B stride
  __shared__ _Float16 pstage[4][16 * 40];  // per-wave P tile, 80B stride
  const int tid  = threadIdx.x;
  const int lane = tid & 31;
  const int wave = tid >> 5;
  const int b = blockIdx.z, h = blockIdx.y;
  const int t0 = (blockIdx.x * 4 + wave) * 16;
  const int rw = lane & 15, grp = lane >> 4;
  const int kbA = grp * 8, kbB = grp * 8;

  const float* qp = qkv + (long)b * CT * 1536 + h * CHD;
  const float* kp = qp + CD;
  const float* vp = qp + 2 * CD;

  // q A-fragments covering d=0..63 (two K=32 chunks): vectorized global loads
  v16h aq[2];
  {
    const float* qrow = qp + (long)(t0 + rw) * 1536;
#pragma unroll
    for (int c = 0; c < 2; ++c) {
      const float4* q0 = (const float4*)(qrow + c * 32 + kbA);
      const float4* q1 = (const float4*)(qrow + c * 32 + 16 + kbA);
      const float4 x0 = q0[0], x1 = q0[1], x2 = q1[0], x3 = q1[1];
      aq[c][0]=(_Float16)x0.x; aq[c][1]=(_Float16)x0.y; aq[c][2]=(_Float16)x0.z; aq[c][3]=(_Float16)x0.w;
      aq[c][4]=(_Float16)x1.x; aq[c][5]=(_Float16)x1.y; aq[c][6]=(_Float16)x1.z; aq[c][7]=(_Float16)x1.w;
      aq[c][8]=(_Float16)x2.x; aq[c][9]=(_Float16)x2.y; aq[c][10]=(_Float16)x2.z; aq[c][11]=(_Float16)x2.w;
      aq[c][12]=(_Float16)x3.x; aq[c][13]=(_Float16)x3.y; aq[c][14]=(_Float16)x3.z; aq[c][15]=(_Float16)x3.w;
    }
  }

  float mrun[8], lsum[8], mv[8];
#pragma unroll
  for (int r = 0; r < 8; ++r) {
    mrun[r] = -1e30f; lsum[r] = 0.f;
    mv[r] = mmult[((long)(b * CH + h)) * CT + t0 + grp * 8 + r];
  }
  v8f oacc[4] = {};
  _Float16* myp = pstage[wave];
  const float* trow = tbl + h * (2 * CT - 1) + (CT - 1) - t0;  // index by (s - row)

  for (int s0 = 0; s0 < CT; s0 += 32) {
    // ---- cooperative K/V chunk staging (coalesced b128) ----
#pragma unroll
    for (int it = 0; it < 4; ++it) {
      const int idx = tid + it * 128;       // [0,512) float4s
      const int n = idx >> 4;               // key 0..31
      const int d = (idx & 15) * 4;         // dim 0..60
      const float4 kv = *(const float4*)(kp + (long)(s0 + n) * 1536 + d);
      Klds[n][d + 0] = (_Float16)kv.x; Klds[n][d + 1] = (_Float16)kv.y;
      Klds[n][d + 2] = (_Float16)kv.z; Klds[n][d + 3] = (_Float16)kv.w;
      const float4 vv = *(const float4*)(vp + (long)(s0 + n) * 1536 + d);
      Vlds[d + 0][n] = (_Float16)vv.x; Vlds[d + 1][n] = (_Float16)vv.y;
      Vlds[d + 2][n] = (_Float16)vv.z; Vlds[d + 3][n] = (_Float16)vv.w;
    }
    __syncthreads();

    // ---- scores: two 16x16 C tiles, K=64 -> 2 WMMAs each; B from LDS ----
    v8f sc[2] = {};
#pragma unroll
    for (int st = 0; st < 2; ++st) {
      const _Float16* krow = &Klds[st * 16 + rw][0];
#pragma unroll
      for (int dc = 0; dc < 2; ++dc) {
        const v8h lo = *(const v8h*)(krow + dc * 32 + kbB);
        const v8h hi = *(const v8h*)(krow + dc * 32 + kbB + 16);
        v16h bf;
#pragma unroll
        for (int j = 0; j < 8; ++j) { bf[j] = lo[j]; bf[j + 8] = hi[j]; }
        sc[st] = WMMA_F32_F16(aq[dc], bf, sc[st]);
      }
    }

    // ---- bias + online softmax (rows r+8*grp live in this lane's group) ----
    float pe[2][8], cmax[8];
#pragma unroll
    for (int r = 0; r < 8; ++r) cmax[r] = -1e30f;
#pragma unroll
    for (int st = 0; st < 2; ++st)
#pragma unroll
      for (int r = 0; r < 8; ++r) {
        const int row = grp * 8 + r;
        const int s = s0 + st * 16 + rw;
        const float v = sc[st][r] * 0.125f + mv[r] * trow[s - row];
        pe[st][r] = v;
        cmax[r] = fmaxf(cmax[r], v);
      }
#pragma unroll
    for (int r = 0; r < 8; ++r)
#pragma unroll
      for (int msk = 1; msk <= 8; msk <<= 1)
        cmax[r] = fmaxf(cmax[r], __shfl_xor(cmax[r], msk));

    float rs[8], ps[8];
#pragma unroll
    for (int r = 0; r < 8; ++r) {
      const float nm = fmaxf(mrun[r], cmax[r]);
      rs[r] = __expf(mrun[r] - nm);
      mrun[r] = nm;
      ps[r] = 0.f;
    }
#pragma unroll
    for (int st = 0; st < 2; ++st)
#pragma unroll
      for (int r = 0; r < 8; ++r) {
        const float e = __expf(pe[st][r] - mrun[r]);
        pe[st][r] = e;
        ps[r] += e;
      }
#pragma unroll
    for (int r = 0; r < 8; ++r) {
#pragma unroll
      for (int msk = 1; msk <= 8; msk <<= 1)
        ps[r] += __shfl_xor(ps[r], msk);
      lsum[r] = lsum[r] * rs[r] + ps[r];
    }
#pragma unroll
    for (int dt = 0; dt < 4; ++dt)
#pragma unroll
      for (int r = 0; r < 8; ++r)
        oacc[dt][r] *= rs[r];

    // ---- stage P (wave-private; LDS in-order within wave, no barrier) ----
#pragma unroll
    for (int st = 0; st < 2; ++st)
#pragma unroll
      for (int r = 0; r < 8; ++r)
        myp[(grp * 8 + r) * 40 + st * 16 + rw] = (_Float16)pe[st][r];
    v16h pf;
    {
      const _Float16* prow = myp + rw * 40;
      const v8h lo = *(const v8h*)(prow + kbA);
      const v8h hi = *(const v8h*)(prow + 16 + kbA);
#pragma unroll
      for (int j = 0; j < 8; ++j) { pf[j] = lo[j]; pf[j + 8] = hi[j]; }
    }

    // ---- P(16x32) @ V(32x64): one WMMA per 16-wide d tile; B from LDS ----
#pragma unroll
    for (int dt = 0; dt < 4; ++dt) {
      const _Float16* vrow = &Vlds[dt * 16 + rw][kbB];
      const v8h lo = *(const v8h*)vrow;
      const v8h hi = *(const v8h*)(vrow + 16);
      v16h vf;
#pragma unroll
      for (int j = 0; j < 8; ++j) { vf[j] = lo[j]; vf[j + 8] = hi[j]; }
      oacc[dt] = WMMA_F32_F16(pf, vf, oacc[dt]);
    }
    __syncthreads();   // protect K/V tiles before next chunk's staging
  }

  // out: (B,T,H*64) token-major so out-proj GEMM reads contiguously
#pragma unroll
  for (int dt = 0; dt < 4; ++dt)
#pragma unroll
    for (int r = 0; r < 8; ++r) {
      const int row = grp * 8 + r;
      out[((long)b * CT + t0 + row) * CD + h * CHD + dt * 16 + rw] = oacc[dt][r] / lsum[r];
    }
}

// ---------------------------------------------------------------------------
// Small kernels
// ---------------------------------------------------------------------------
__global__ void transpose_to_tokens(const float* __restrict__ x, float* __restrict__ y)
{ // x (B,D,T) -> y (B,T,D)
  long i = (long)blockIdx.x * 256 + threadIdx.x;
  if (i >= (long)CB * CD * CT) return;
  int t = (int)(i % CT); int d = (int)((i / CT) % CD); int b = (int)(i / ((long)CD * CT));
  y[((long)b * CT + t) * CD + d] = x[i];
}

__global__ void transpose_to_channels(const float* __restrict__ y, float* __restrict__ x)
{ // y (B,T,D) -> x (B,D,T)
  long i = (long)blockIdx.x * 256 + threadIdx.x;
  if (i >= (long)CB * CT * CD) return;
  int d = (int)(i % CD); int t = (int)((i / CD) % CT); int b = (int)(i / ((long)CT * CD));
  x[((long)b * CD + d) * CT + t] = y[i];
}

__global__ void mgate_kernel(const float* __restrict__ qkv, const float* __restrict__ gu,
                             const float* __restrict__ gw, const float* __restrict__ gs,
                             float* __restrict__ mm)
{ // m[b,h,t] = 1 + gu + (1-gu)*scale*gr
  int i = blockIdx.x * 256 + threadIdx.x;
  if (i >= CB * CH * CT) return;
  int t = i % CT; int h = (i / CT) % CH; int b = i / (CT * CH);
  const float* q = qkv + ((long)b * CT + t) * 1536 + h * CHD;
  float du = 0.f, dv = 0.f;
#pragma unroll 8
  for (int d = 0; d < CHD; ++d) { du += q[d] * gu[h * CHD + d]; dv += q[d] * gw[h * CHD + d]; }
  const float sgu = 1.f / (1.f + __expf(-du));
  const float sgr = 1.f / (1.f + __expf(-dv));
  mm[i] = 1.f + sgu + (1.f - sgu) * gs[h] * sgr;
}

__global__ void relbias_kernel(const float* __restrict__ E, float* __restrict__ tbl)
{ // tbl[h, rel+T-1] = rel_embed[bucket(rel), h]
  int i = blockIdx.x * 256 + threadIdx.x;
  const int P = 2 * CT - 1;
  if (i >= CH * P) return;
  int h = i / P, p = i % P;
  int rel = p - (CT - 1);
  int sgn = (rel >= 0) ? 1 : 0;
  int ab = rel < 0 ? -rel : rel;
  int bucket;
  if (ab < 80) bucket = ab;
  else {
    float lr = logf((float)ab * (1.0f / 80.f)) / logf(10.f);  // log(MAXD/st)=log(10)
    int lp = 80 + (int)(lr * 80.f);
    bucket = lp < 159 ? lp : 159;
  }
  bucket += sgn * 160;
  if (bucket < 0) bucket = 0; if (bucket > 319) bucket = 319;
  tbl[i] = E[bucket * CH + h];
}

__global__ void zero_stats_kernel(float* __restrict__ p)
{ if (threadIdx.x < 4) p[threadIdx.x] = 0.f; }

__global__ void __launch_bounds__(256)
gn_stats_kernel(const float* __restrict__ x, float* __restrict__ stats)
{
  __shared__ float ss[256], sq[256];
  const long n = (long)CD * CT;
  const int b = blockIdx.y;
  const float* xb = x + (long)b * n;
  float s = 0.f, q = 0.f;
  for (long i = (long)blockIdx.x * blockDim.x + threadIdx.x; i < n;
       i += (long)gridDim.x * blockDim.x) {
    float v = xb[i]; s += v; q += v * v;
  }
  ss[threadIdx.x] = s; sq[threadIdx.x] = q;
  __syncthreads();
  for (int o = 128; o > 0; o >>= 1) {
    if ((int)threadIdx.x < o) { ss[threadIdx.x] += ss[threadIdx.x + o]; sq[threadIdx.x] += sq[threadIdx.x + o]; }
    __syncthreads();
  }
  if (threadIdx.x == 0) { atomicAdd(&stats[b * 2], ss[0]); atomicAdd(&stats[b * 2 + 1], sq[0]); }
}

__global__ void gn_apply_kernel(const float* __restrict__ x, const float* __restrict__ stats,
                                const float* __restrict__ g, const float* __restrict__ bt,
                                float* __restrict__ y, int silu_mode)
{
  long i = (long)blockIdx.x * 256 + threadIdx.x;
  if (i >= (long)CB * CT * CD) return;
  int d = (int)(i % CD);
  int b = (int)(i / ((long)CT * CD));
  const float inv = 1.0f / ((float)CD * (float)CT);
  const float mean = stats[b * 2] * inv;
  const float var  = stats[b * 2 + 1] * inv - mean * mean;
  float v = (x[i] - mean) * rsqrtf(var + 1e-5f) * g[d] + bt[d];
  if (silu_mode) v = v / (1.f + __expf(-v));
  y[i] = v;
}

__global__ void glu_kernel(const float* __restrict__ hg, float* __restrict__ y)
{
  long i = (long)blockIdx.x * 256 + threadIdx.x;
  if (i >= (long)CB * CT * CD) return;
  int d = (int)(i % CD); long bt = i / CD;
  const float a  = hg[bt * (2 * CD) + d];
  const float gv = hg[bt * (2 * CD) + CD + d];
  y[i] = a / (1.f + __expf(-gv));
}

__global__ void dwconv_kernel(const float* __restrict__ gl, const float* __restrict__ w,
                              const float* __restrict__ bsc, float* __restrict__ y)
{
  long i = (long)blockIdx.x * 256 + threadIdx.x;
  if (i >= (long)CB * CT * CD) return;
  int d = (int)(i % CD); int t = (int)((i / CD) % CT); int b = (int)(i / ((long)CT * CD));
  const float* base = gl + (long)b * CT * CD + d;
  float s = bsc[d];
#pragma unroll
  for (int k = 0; k < 31; ++k) {
    const int tt = t + k - 15;
    if (tt >= 0 && tt < CT) s += base[(long)tt * CD] * w[d * 31 + k];
  }
  y[i] = s;
}

// ---------------------------------------------------------------------------
extern "C" void kernel_launch(void* const* d_in, const int* in_sizes, int n_in,
                              void* d_out, int out_size, void* d_ws, size_t ws_size,
                              hipStream_t stream)
{
  const float* x       = (const float*)d_in[0];
  const float* ff1_w1  = (const float*)d_in[1];
  const float* ff1_b1  = (const float*)d_in[2];
  const float* ff1_w2  = (const float*)d_in[3];
  const float* ff1_b2  = (const float*)d_in[4];
  const float* qkv_w   = (const float*)d_in[5];
  const float* qkv_b   = (const float*)d_in[6];
  const float* out_w   = (const float*)d_in[7];
  const float* out_b   = (const float*)d_in[8];
  const float* gn1_g   = (const float*)d_in[9];
  const float* gn1_b   = (const float*)d_in[10];
  const float* pw1_w   = (const float*)d_in[11];
  const float* pw1_b   = (const float*)d_in[12];
  const float* dw_w    = (const float*)d_in[13];
  const float* dw_b    = (const float*)d_in[14];
  const float* gn2_g   = (const float*)d_in[15];
  const float* gn2_b   = (const float*)d_in[16];
  const float* pw2_w   = (const float*)d_in[17];
  const float* pw2_b   = (const float*)d_in[18];
  const float* ff2_w1  = (const float*)d_in[19];
  const float* ff2_b1  = (const float*)d_in[20];
  const float* ff2_w2  = (const float*)d_in[21];
  const float* ff2_b2  = (const float*)d_in[22];
  const float* rel_e   = (const float*)d_in[23];
  const float* gate_u  = (const float*)d_in[24];
  const float* gate_wv = (const float*)d_in[25];
  const float* gate_sc = (const float*)d_in[26];
  float* out = (float*)d_out;
  (void)in_sizes; (void)n_in; (void)out_size; (void)ws_size;

  const size_t NTOK = (size_t)CB * CT;      // 4096
  float* S0  = (float*)d_ws;                // 8 MB  (B,T,D)
  float* S1  = S0 + NTOK * CD;              // 8 MB
  float* S2  = S1 + NTOK * CD;              // 8 MB
  float* H1  = S2 + NTOK * CD;              // 32 MB (B,T,4D)
  float* QK  = H1 + NTOK * CD * 4;          // 24 MB (B,T,3D) / reused as (B,T,2D)
  float* MM  = QK + NTOK * CD * 3;          // (B,H,T)
  float* TBL = MM + (size_t)CB * CH * CT;   // (H, 2T-1)
  float* ST  = TBL + (size_t)CH * (2 * CT - 1);  // 4 stats floats

  const int M = (int)NTOK;
  const int EW = (int)((NTOK * CD + 255) / 256);   // elementwise grid

  auto gemm_n = [&](const float* A, const float* Bm, long bK,
                    const float* bias, const float* res, float* C,
                    int N, int K, float alpha, int act) {
    dim3 grid(N / 64, M / 128);
    wmma_gemm_kernel<0><<<grid, 256, 0, stream>>>(A, Bm, bK, 1, bias, res, C,
                                                  M, N, K, alpha, act);
  };
  auto gemm_t = [&](const float* A, const float* Bm, long bN,
                    const float* bias, const float* res, float* C,
                    int N, int K, float alpha, int act) {
    dim3 grid(N / 64, M / 128);
    wmma_gemm_kernel<1><<<grid, 256, 0, stream>>>(A, Bm, 1, bN, bias, res, C,
                                                  M, N, K, alpha, act);
  };

  // 1. s0 = x^T (token-major)
  transpose_to_tokens<<<EW, 256, 0, stream>>>(x, S0);
  // 2-3. FFN1: s1 = s0 + 0.5*(gelu(s0@W1+b1)@W2+b2)
  gemm_n(S0, ff1_w1, 2048, ff1_b1, nullptr, H1, 2048, 512, 1.0f, 1);
  gemm_n(H1, ff1_w2, 512, ff1_b2, S0, S1, 512, 2048, 0.5f, 0);
  // 4. qkv = s1 @ qkv_w + b
  gemm_n(S1, qkv_w, 1536, qkv_b, nullptr, QK, 1536, 512, 1.0f, 0);
  // 5-6. gate multipliers + rel-bias table
  mgate_kernel<<<(CB * CH * CT + 255) / 256, 256, 0, stream>>>(QK, gate_u, gate_wv, gate_sc, MM);
  relbias_kernel<<<(CH * (2 * CT - 1) + 255) / 256, 256, 0, stream>>>(rel_e, TBL);
  // 7. attention -> S0 (token-major, heads concatenated)
  flash_attn_kernel<<<dim3(CT / 64, CH, CB), 128, 0, stream>>>(QK, MM, TBL, S0);
  // 8. s2 = s1 + attn @ out_w + out_b
  gemm_n(S0, out_w, 512, out_b, S1, S2, 512, 512, 1.0f, 0);
  // 9. GroupNorm1 -> S0
  zero_stats_kernel<<<1, 32, 0, stream>>>(ST);
  gn_stats_kernel<<<dim3(256, CB), 256, 0, stream>>>(S2, ST);
  gn_apply_kernel<<<EW, 256, 0, stream>>>(S2, ST, gn1_g, gn1_b, S0, 0);
  // 10. pw1 (B read transposed: B[k,n]=W[n,k], contiguous in k) -> QK (B,T,1024)
  gemm_t(S0, pw1_w, 512, pw1_b, nullptr, QK, 1024, 512, 1.0f, 0);
  // 11. GLU -> S1
  glu_kernel<<<EW, 256, 0, stream>>>(QK, S1);
  // 12. depthwise conv K=31 -> S0
  dwconv_kernel<<<EW, 256, 0, stream>>>(S1, dw_w, dw_b, S0);
  // 13. GroupNorm2 + SiLU -> S1
  zero_stats_kernel<<<1, 32, 0, stream>>>(ST);
  gn_stats_kernel<<<dim3(256, CB), 256, 0, stream>>>(S0, ST);
  gn_apply_kernel<<<EW, 256, 0, stream>>>(S0, ST, gn2_g, gn2_b, S1, 1);
  // 14. c2 = s2 + pw2(hn2) -> S0
  gemm_t(S1, pw2_w, 512, pw2_b, S2, S0, 512, 512, 1.0f, 0);
  // 15-16. FFN2 -> S2
  gemm_n(S0, ff2_w1, 2048, ff2_b1, nullptr, H1, 2048, 512, 1.0f, 1);
  gemm_n(H1, ff2_w2, 512, ff2_b2, S0, S2, 512, 2048, 0.5f, 0);
  // 17. output (B,D,T)
  transpose_to_channels<<<EW, 256, 0, stream>>>(S2, out);
}
